// Decoder_57449482551830
// MI455X (gfx1250) — matrix operand order repaired and checked by reading, not compiled, not run on hardware
//
#include <hip/hip_runtime.h>

// ---------------------------------------------------------------------------
// MI455X (gfx1250) decoder: embed -> 2x LSTM -> global attention.
// All GEMMs use v_wmma_f32_16x16x32_bf16. Big GEMMs use a double-buffered
// GLOBAL_LOAD_ASYNC_TO_LDS_B128 pipeline (ASYNCcnt) when the toolchain
// exposes the builtins. The LSTM scan is a persistent kernel (4 WGs, one per
// 16-row batch slice): c in VGPRs, h ping-pong in LDS, no cross-WG sync.
// d_out layout (f32): out (32*64*1024) | hn (2*64*1024) | cn (2*64*1024) | attn_last (64*64)
// ---------------------------------------------------------------------------

#define DEV __device__ __forceinline__

typedef __attribute__((ext_vector_type(16))) __bf16        v16bf;
typedef __attribute__((ext_vector_type(8)))  float         v8f;
typedef __attribute__((ext_vector_type(4)))  unsigned int  u32x4;
typedef __attribute__((ext_vector_type(2)))  unsigned int  u32x2;
typedef __attribute__((ext_vector_type(4)))  int           i32x4;
typedef __attribute__((ext_vector_type(4)))  float         f32x4;

union Frag { u32x4 q[2]; v16bf v; };

#if defined(__HIP_DEVICE_COMPILE__) && \
    __has_builtin(__builtin_amdgcn_global_load_async_to_lds_b128)
#define HAS_ASYNC_LDS 1
#else
#define HAS_ASYNC_LDS 0
#endif

DEV unsigned short f2bf(float f) {
  unsigned int u = __float_as_uint(f);
  u += 0x7FFFu + ((u >> 16) & 1u);           // round-to-nearest-even
  return (unsigned short)(u >> 16);
}
DEV float sigm(float x) { return 1.0f / (1.0f + __expf(-x)); }

DEV v8f wmma_bf16(const Frag& a, const Frag& b, v8f c) {
#if defined(__HIP_DEVICE_COMPILE__)
  return __builtin_amdgcn_wmma_f32_16x16x32_bf16(false, a.v, false, b.v,
                                                 (short)0, c, false, false);
#else
  (void)a; (void)b; return c;
#endif
}

// async global->LDS 128-bit copy (per-lane), tracked by ASYNCcnt
DEV void async_copy_b128(const unsigned short* g, unsigned short* l) {
#if HAS_ASYNC_LDS
  __builtin_amdgcn_global_load_async_to_lds_b128(
      (__attribute__((address_space(1))) i32x4*)(unsigned short*)g,
      (__attribute__((address_space(3))) i32x4*)l, 0, 0);
#else
  *(u32x4*)l = *(const u32x4*)g;
#endif
}

DEV void wait_async_all() {
#if defined(__HIP_DEVICE_COMPILE__)
#if __has_builtin(__builtin_amdgcn_s_wait_asynccnt)
  __builtin_amdgcn_s_wait_asynccnt(0);
#elif HAS_ASYNC_LDS
  asm volatile("s_wait_asynccnt 0x0" ::: "memory");
#endif
#endif
}

// A fragment: 16x32 bf16 tile, row-major source rows (lane&15 = M row).
// ISA layout: lanes0-15 hold K0..7 & K16..23, lanes16-31 hold K8..15 & K24..31.
DEV Frag load_fragA(const unsigned short* p, size_t stride) {
  const int l = threadIdx.x & 31;
  const unsigned short* r = p + (size_t)(l & 15) * stride + ((l >> 4) << 3);
  Frag f;
  f.q[0] = *(const u32x4*)(r);
  f.q[1] = *(const u32x4*)(r + 16);
  return f;
}

// B fragment: 32x16 bf16 tile; source rows are the 16 N-columns with K
// contiguous (i.e. B^T in memory). lanes0-15 hold K0..7 & K16..23, lanes16-31 K8..15 & K24..31.
DEV Frag load_fragB(const unsigned short* p, size_t stride) {
  const int l = threadIdx.x & 31;
  const unsigned short* r = p + (size_t)(l & 15) * stride + ((l >> 4) << 3);
  Frag f;
  f.q[0] = *(const u32x4*)(r);
  f.q[1] = *(const u32x4*)(r + 16);
  return f;
}

// ---------------------------------------------------------------------------
// elementwise helpers
// ---------------------------------------------------------------------------
__global__ void cvt4_kernel(const float* __restrict__ src,
                            unsigned short* __restrict__ dst, int n4) {
  int i = blockIdx.x * blockDim.x + threadIdx.x;
  if (i < n4) {
    f32x4 f = ((const f32x4*)src)[i];
    u32x2 o;
    o.x = (unsigned)f2bf(f.x) | ((unsigned)f2bf(f.y) << 16);
    o.y = (unsigned)f2bf(f.z) | ((unsigned)f2bf(f.w) << 16);
    ((u32x2*)dst)[i] = o;
  }
}

__global__ void bias_sum_kernel(const float* __restrict__ a,
                                const float* __restrict__ b,
                                float* __restrict__ out, int n) {
  int i = blockIdx.x * blockDim.x + threadIdx.x;
  if (i < n) out[i] = a[i] + b[i];
}

// embedding gather (padding_idx = 0 -> zero row), output bf16 (T*B, D)
__global__ void embed_kernel(const int* __restrict__ idx,
                             const float* __restrict__ emb,
                             unsigned short* __restrict__ xbf) {
  const int row = blockIdx.x;
  const int id  = idx[row];
  const float* src = emb + (size_t)id * 1024;
  unsigned short* dst = xbf + (size_t)row * 1024;
  for (int j = threadIdx.x; j < 1024; j += blockDim.x)
    dst[j] = (id == 0) ? (unsigned short)0 : f2bf(src[j]);
}

// copy layer-1 hidden (bf16) into right half of concat buffer
__global__ void cat_copy_kernel(const unsigned short* __restrict__ ybf,
                                unsigned short* __restrict__ catbf) {
  int i = blockIdx.x * blockDim.x + threadIdx.x;   // 0..2048*1024
  int row = i >> 10, d = i & 1023;
  catbf[(size_t)row * 2048 + 1024 + d] = ybf[i];
}

// ---------------------------------------------------------------------------
// Generic WMMA GEMM: C(MxN) = A(MxK,bf16) * B(NxK,bf16)^T + bias, opt tanh.
// Workgroup tile 64x128, 8 waves (4 M-subtiles x 2 N-halves).
// Double-buffered async global->LDS staging: issue copies for tile k+1 after
// the barrier, WMMA on tile k while they fly. grid=(N/128, M/64), block=256.
// ---------------------------------------------------------------------------
__global__ __launch_bounds__(256) void gemm_bt_kernel(
    const unsigned short* __restrict__ A, int lda,
    const unsigned short* __restrict__ B, int ldb,
    const float* __restrict__ bias,
    float* __restrict__ Cf, unsigned short* __restrict__ Cb, int ldc,
    int K, int act) {
  __shared__ __align__(16) unsigned short As[2][64 * 40];
  __shared__ __align__(16) unsigned short Bs[2][128 * 40];
  const int m0 = blockIdx.y * 64;
  const int n0 = blockIdx.x * 128;
  const int wave = threadIdx.x >> 5;
  const int lane = threadIdx.x & 31;
  const int mt = wave & 3;        // M sub-tile
  const int nh = wave >> 2;       // N half (4 tiles each)
  v8f acc[4] = {v8f{}, v8f{}, v8f{}, v8f{}};

  const int ra = threadIdx.x >> 2;            // A stage: 64 rows x 1 b128
  const int ca = (threadIdx.x & 3) << 3;
  auto issue_tile = [&](int k0, int buf) {
    async_copy_b128(A + (size_t)(m0 + ra) * lda + k0 + ca,
                    &As[buf][ra * 40 + ca]);
#pragma unroll
    for (int p = 0; p < 2; ++p) {             // B stage: 128 rows x 2 passes
      int t = threadIdx.x + (p << 8);
      int r = t >> 2, c = (t & 3) << 3;
      async_copy_b128(B + (size_t)(n0 + r) * ldb + k0 + c,
                      &Bs[buf][r * 40 + c]);
    }
  };

  issue_tile(0, 0);
  int buf = 0;
  for (int k0 = 0; k0 < K; k0 += 32) {
    wait_async_all();
    __syncthreads();                          // tile `buf` fully in LDS
    if (k0 + 32 < K) issue_tile(k0 + 32, buf ^ 1);
    Frag a = load_fragA(&As[buf][mt * 16 * 40], 40);
    Frag bfr[4];
#pragma unroll
    for (int j = 0; j < 4; ++j)
      bfr[j] = load_fragB(&Bs[buf][(nh * 64 + j * 16) * 40], 40);
#pragma unroll
    for (int j = 0; j < 4; ++j)
      acc[j] = wmma_bf16(a, bfr[j], acc[j]);
    buf ^= 1;
  }
#pragma unroll
  for (int j = 0; j < 4; ++j) {
    int n = n0 + nh * 64 + j * 16 + (lane & 15);
    float bv = bias ? bias[n] : 0.0f;
#pragma unroll
    for (int r = 0; r < 8; ++r) {
      int m = m0 + mt * 16 + r + ((lane >> 4) << 3);
      float v = acc[j][r] + bv;
      if (act) v = tanhf(v);
      if (Cf) Cf[(size_t)m * ldc + n] = v;
      if (Cb) Cb[(size_t)m * ldc + n] = f2bf(v);
    }
  }
}

// ---------------------------------------------------------------------------
// Persistent LSTM scan for one layer. grid = 4 (batch slices of 16), block 256.
// xg holds precomputed x@Wih^T + (bih+bhh) for all timesteps.
// Each wave owns 8 d-tiles; per step computes 4 gate tiles (i,f,g,o) via
// WMMA over K=1024, applies the cell nonlinearity on the accumulators,
// keeps c in VGPRs, ping-pongs bf16 h through LDS.
// ---------------------------------------------------------------------------
__global__ __launch_bounds__(256) void lstm_layer_kernel(
    const float* __restrict__ xg,             // (T*B, 4096)
    const unsigned short* __restrict__ whhb,  // (4096, 1024) bf16
    const float* __restrict__ h0,             // (B, 1024) layer slice
    const float* __restrict__ c0,             // (B, 1024) layer slice
    unsigned short* __restrict__ ybf,         // (T*B, 1024) bf16 out
    float* __restrict__ hn, float* __restrict__ cn) {  // (B, 1024)
  __shared__ __align__(16) unsigned short Hs[2][16 * 1032];
  const int b0 = blockIdx.x * 16;
  const int wave = threadIdx.x >> 5;
  const int lane = threadIdx.x & 31;
  const int nloc = lane & 15;
  const int moff = (lane >> 4) << 3;

  // init h (bf16 -> LDS buffer 0)
  for (int i = threadIdx.x; i < 16 * 1024; i += 256) {
    int r = i >> 10, c = i & 1023;
    Hs[0][r * 1032 + c] = f2bf(h0[(size_t)(b0 + r) * 1024 + c]);
  }
  // init c (f32 -> accumulator-layout registers)
  v8f creg[8];
#pragma unroll
  for (int jj = 0; jj < 8; ++jj) {
    int d = (wave * 8 + jj) * 16 + nloc;
#pragma unroll
    for (int r = 0; r < 8; ++r)
      creg[jj][r] = c0[(size_t)(b0 + r + moff) * 1024 + d];
  }
  __syncthreads();

  for (int t = 0; t < 32; ++t) {
    const unsigned short* hread = Hs[t & 1];
    unsigned short* hwrite = Hs[(t + 1) & 1];
    const float* xgt = xg + (size_t)(t * 64 + b0) * 4096;
    for (int jj = 0; jj < 8; ++jj) {
      const int dbase = (wave * 8 + jj) * 16;
      v8f g[4] = {v8f{}, v8f{}, v8f{}, v8f{}};
      for (int k0 = 0; k0 < 1024; k0 += 32) {
        Frag a = load_fragA(hread + k0, 1032);
        Frag bfr[4];
#pragma unroll
        for (int gi = 0; gi < 4; ++gi)
          bfr[gi] = load_fragB(
              whhb + (size_t)(gi * 1024 + dbase) * 1024 + k0, 1024);
#pragma unroll
        for (int gi = 0; gi < 4; ++gi) g[gi] = wmma_bf16(a, bfr[gi], g[gi]);
      }
      const int d = dbase + nloc;
#pragma unroll
      for (int r = 0; r < 8; ++r) {
        const int brow = r + moff;
        const float* xr = xgt + (size_t)brow * 4096;
        float iv = sigm(g[0][r] + xr[d]);
        float fv = sigm(g[1][r] + xr[1024 + d]);
        float gv = tanhf(g[2][r] + xr[2048 + d]);
        float ov = sigm(g[3][r] + xr[3072 + d]);
        float cv = fv * creg[jj][r] + iv * gv;
        float hv = ov * tanhf(cv);
        creg[jj][r] = cv;
        unsigned short hb = f2bf(hv);
        hwrite[brow * 1032 + d] = hb;
        ybf[(size_t)(t * 64 + b0 + brow) * 1024 + d] = hb;
        if (t == 31) {
          hn[(size_t)(b0 + brow) * 1024 + d] = hv;
          cn[(size_t)(b0 + brow) * 1024 + d] = cv;
        }
      }
    }
    __syncthreads();
  }
}

// ---------------------------------------------------------------------------
// Fused attention scores + softmax. grid = (T/16, B), block = 128 (4 waves).
// Wave w computes the 16x16 tile (t0..t0+15) x (s = 16w..16w+15) over K=1024,
// then 16 threads do the row softmax from LDS. Writes bf16 attn (T*B,S) and
// the t=31 rows (f32) straight into d_out.
// ---------------------------------------------------------------------------
__global__ __launch_bounds__(128) void attn_scores_kernel(
    const unsigned short* __restrict__ qbf,    // (T*B, D)
    const unsigned short* __restrict__ ctxbf,  // (S*B, D), row = s*B+b
    unsigned short* __restrict__ attnbf,       // (T*B, S)
    float* __restrict__ attn_last) {           // (B, S)
  __shared__ float Ss[16 * 68];
  const int t0 = blockIdx.x * 16;
  const int b = blockIdx.y;
  const int wave = threadIdx.x >> 5;
  const int lane = threadIdx.x & 31;
  v8f acc = {};
  for (int k0 = 0; k0 < 1024; k0 += 32) {
    Frag a = load_fragA(qbf + ((size_t)(t0 * 64 + b)) * 1024 + k0,
                        (size_t)64 * 1024);
    Frag bb = load_fragB(ctxbf + ((size_t)(wave * 16) * 64 + b) * 1024 + k0,
                         (size_t)64 * 1024);
    acc = wmma_bf16(a, bb, acc);
  }
#pragma unroll
  for (int r = 0; r < 8; ++r)
    Ss[(r + ((lane >> 4) << 3)) * 68 + wave * 16 + (lane & 15)] = acc[r];
  __syncthreads();
  if (threadIdx.x < 16) {
    const int row = threadIdx.x;
    float mx = -3.4e38f;
    for (int s = 0; s < 64; ++s) mx = fmaxf(mx, Ss[row * 68 + s]);
    float sum = 0.0f;
    for (int s = 0; s < 64; ++s) {
      float e = __expf(Ss[row * 68 + s] - mx);
      Ss[row * 68 + s] = e;
      sum += e;
    }
    const float inv = 1.0f / sum;
    const int t = t0 + row;
    for (int s = 0; s < 64; ++s) {
      float a = Ss[row * 68 + s] * inv;
      attnbf[((size_t)t * 64 + b) * 64 + s] = f2bf(a);
      if (t == 31) attn_last[b * 64 + s] = a;
    }
  }
}

// ---------------------------------------------------------------------------
// weighted = attn @ ctx (per-batch, M=32,K=64,N=1024); ctx tile is staged
// transposed through LDS so B fragments are two b128 LDS reads. Result goes
// bf16 straight into the LEFT half of the concat buffer.
// grid = (N/64, T/16, B), block = 128 (4 waves, one 16-wide n-tile each).
// ---------------------------------------------------------------------------
__global__ __launch_bounds__(128) void attn_weighted_kernel(
    const unsigned short* __restrict__ attnbf,  // (T*B, S)
    const unsigned short* __restrict__ ctxbf,   // (S*B, D)
    unsigned short* __restrict__ catbf) {       // (T*B, 2D)
  __shared__ __align__(16) unsigned short Bt[64 * 72];  // [n][k]
  const int n0 = blockIdx.x * 64;
  const int mt = blockIdx.y;
  const int b = blockIdx.z;
  for (int i = threadIdx.x; i < 64 * 16; i += 128) {
    int s = i >> 4, nq = (i & 15) << 2;
    const unsigned short* src = ctxbf + ((size_t)s * 64 + b) * 1024 + n0 + nq;
#pragma unroll
    for (int u = 0; u < 4; ++u) Bt[(nq + u) * 72 + s] = src[u];
  }
  __syncthreads();
  const int wave = threadIdx.x >> 5;
  const int lane = threadIdx.x & 31;
  v8f acc = {};
#pragma unroll
  for (int k0 = 0; k0 < 64; k0 += 32) {
    Frag a = load_fragA(attnbf + ((size_t)(mt * 16) * 64 + b) * 64 + k0,
                        (size_t)64 * 64);
    Frag bb = load_fragB(Bt + (wave * 16) * 72 + k0, 72);
    acc = wmma_bf16(a, bb, acc);
  }
#pragma unroll
  for (int r = 0; r < 8; ++r) {
    int t = mt * 16 + r + ((lane >> 4) << 3);
    int d = n0 + wave * 16 + (lane & 15);
    catbf[((size_t)t * 64 + b) * 2048 + d] = f2bf(acc[r]);
  }
}

// ---------------------------------------------------------------------------
// host launcher
// ---------------------------------------------------------------------------
extern "C" void kernel_launch(void* const* d_in, const int* in_sizes, int n_in,
                              void* d_out, int out_size, void* d_ws,
                              size_t ws_size, hipStream_t stream) {
  (void)in_sizes; (void)n_in; (void)out_size; (void)ws_size;
  const int D = 1024, TB = 2048, Bb = 64;

  const int*   inp   = (const int*)d_in[0];
  const float* h0    = (const float*)d_in[1];
  const float* c0    = (const float*)d_in[2];
  const float* ctx   = (const float*)d_in[3];
  const float* emb   = (const float*)d_in[5];
  const float* wih0  = (const float*)d_in[6];
  const float* whh0  = (const float*)d_in[7];
  const float* bih0  = (const float*)d_in[8];
  const float* bhh0  = (const float*)d_in[9];
  const float* wih1  = (const float*)d_in[10];
  const float* whh1  = (const float*)d_in[11];
  const float* bih1  = (const float*)d_in[12];
  const float* bhh1  = (const float*)d_in[13];
  const float* w_in  = (const float*)d_in[14];
  const float* w_out = (const float*)d_in[15];

  char* ws = (char*)d_ws;
  size_t off = 0;
  auto alloc = [&](size_t bytes) -> void* {
    void* p = ws + off;
    off += (bytes + 255) & ~(size_t)255;
    return p;
  };
  unsigned short* wih0b = (unsigned short*)alloc((size_t)4 * D * D * 2);
  unsigned short* whh0b = (unsigned short*)alloc((size_t)4 * D * D * 2);
  unsigned short* wih1b = (unsigned short*)alloc((size_t)4 * D * D * 2);
  unsigned short* whh1b = (unsigned short*)alloc((size_t)4 * D * D * 2);
  unsigned short* winb  = (unsigned short*)alloc((size_t)D * D * 2);
  unsigned short* woutb = (unsigned short*)alloc((size_t)D * 2 * D * 2);
  unsigned short* ctxb  = (unsigned short*)alloc((size_t)64 * Bb * D * 2);
  unsigned short* x0b   = (unsigned short*)alloc((size_t)TB * D * 2);
  unsigned short* y0b   = (unsigned short*)alloc((size_t)TB * D * 2);
  unsigned short* y1b   = (unsigned short*)alloc((size_t)TB * D * 2);
  unsigned short* qb    = (unsigned short*)alloc((size_t)TB * D * 2);
  unsigned short* attnb = (unsigned short*)alloc((size_t)TB * 64 * 2);
  unsigned short* catb  = (unsigned short*)alloc((size_t)TB * 2 * D * 2);
  float* xg    = (float*)alloc((size_t)TB * 4 * D * 4);
  float* bsum0 = (float*)alloc((size_t)4 * D * 4);
  float* bsum1 = (float*)alloc((size_t)4 * D * 4);

  float* outp  = (float*)d_out;
  float* hn    = outp + (size_t)TB * D;                // (2,64,1024)
  float* cn    = hn + (size_t)2 * Bb * D;
  float* attnL = cn + (size_t)2 * Bb * D;              // (64,64)

  auto cvt = [&](const float* s, unsigned short* d, int n) {
    int n4 = n >> 2;
    cvt4_kernel<<<(n4 + 255) / 256, 256, 0, stream>>>(s, d, n4);
  };
  // weight / context conversions (f32 -> bf16)
  cvt(wih0, wih0b, 4 * D * D);
  cvt(whh0, whh0b, 4 * D * D);
  cvt(wih1, wih1b, 4 * D * D);
  cvt(whh1, whh1b, 4 * D * D);
  cvt(w_in, winb, D * D);
  cvt(w_out, woutb, D * 2 * D);
  cvt(ctx, ctxb, 64 * Bb * D);
  bias_sum_kernel<<<16, 256, 0, stream>>>(bih0, bhh0, bsum0, 4 * D);
  bias_sum_kernel<<<16, 256, 0, stream>>>(bih1, bhh1, bsum1, 4 * D);

  // embedding (padding_idx 0) -> bf16
  embed_kernel<<<TB, 256, 0, stream>>>(inp, emb, x0b);

  // layer 0: xg0 = x @ wih0^T + b0 ; persistent scan
  gemm_bt_kernel<<<dim3(32, 32), 256, 0, stream>>>(
      x0b, D, wih0b, D, bsum0, xg, nullptr, 4 * D, D, 0);
  lstm_layer_kernel<<<4, 256, 0, stream>>>(xg, whh0b, h0, c0, y0b, hn, cn);

  // layer 1
  gemm_bt_kernel<<<dim3(32, 32), 256, 0, stream>>>(
      y0b, D, wih1b, D, bsum1, xg, nullptr, 4 * D, D, 0);
  lstm_layer_kernel<<<4, 256, 0, stream>>>(xg, whh1b, h0 + (size_t)Bb * D,
                                           c0 + (size_t)Bb * D, y1b,
                                           hn + (size_t)Bb * D,
                                           cn + (size_t)Bb * D);

  // attention: q = y1 @ w_in^T (bf16 out)
  gemm_bt_kernel<<<dim3(8, 32), 256, 0, stream>>>(
      y1b, D, winb, D, nullptr, nullptr, qb, D, D, 0);
  attn_scores_kernel<<<dim3(2, 64), 128, 0, stream>>>(qb, ctxb, attnb, attnL);
  attn_weighted_kernel<<<dim3(16, 2, 64), 128, 0, stream>>>(attnb, ctxb, catb);
  cat_copy_kernel<<<8192, 256, 0, stream>>>(y1b, catb);

  // out = tanh(cat @ w_out^T) -> f32 d_out
  gemm_bt_kernel<<<dim3(8, 32), 256, 0, stream>>>(
      catb, 2 * D, woutb, 2 * D, nullptr, outp, nullptr, D, 2 * D, 1);
}
